// MultiHeadAttention_11793980195413
// MI455X (gfx1250) — matrix operand order repaired and checked by reading, not compiled, not run on hardware
//
#include <hip/hip_runtime.h>
#include <hip/hip_bf16.h>
#include <math.h>

// ---------------------------------------------------------------------------
// MI455X (gfx1250) multi-head attention, bf16 WMMA path, flash-attention style
// ---------------------------------------------------------------------------

typedef __attribute__((ext_vector_type(16))) __bf16 v16bf;
typedef __attribute__((ext_vector_type(8)))  __bf16 v8bf;
typedef __attribute__((ext_vector_type(8)))  float  v8f;
typedef __attribute__((ext_vector_type(4)))  unsigned int u32x4;
typedef __attribute__((ext_vector_type(8)))  int i32x8;
typedef __attribute__((ext_vector_type(4)))  int i32x4;

union BF16x16 { v16bf v; v8bf h[2]; };

#define L2E 1.4426950408889634f

// Tensor Data Mover: 5-arg builtin on ROCm 7.2 / clang-22 (probe-verified).
#if defined(__HIP_DEVICE_COMPILE__)
#  if __has_builtin(__builtin_amdgcn_tensor_load_to_lds) && \
      __has_builtin(__builtin_amdgcn_s_wait_tensorcnt) && \
      (__clang_major__ <= 22)
#    define USE_TDM 1
#  else
#    define USE_TDM 0
#  endif
#else
#  define USE_TDM 0
#endif

static __device__ __forceinline__ v8f vzero8() {
    v8f z = {0.f,0.f,0.f,0.f,0.f,0.f,0.f,0.f};
    return z;
}

static __device__ __forceinline__ v8bf cvt8(float4 a, float4 b) {
    v8bf r;
    r[0] = (__bf16)a.x; r[1] = (__bf16)a.y; r[2] = (__bf16)a.z; r[3] = (__bf16)a.w;
    r[4] = (__bf16)b.x; r[5] = (__bf16)b.y; r[6] = (__bf16)b.z; r[7] = (__bf16)b.w;
    return r;
}

// ---------------------------------------------------------------------------
// GEMM: C[M x N] = A[M x 1024] * W[N x 1024]^T + bias
//   MODE 0: QKV epilogue -> scatter bf16 Q/K/V into [B,H,S,HD] (Q pre-scaled 1/8)
//   MODE 1: proj epilogue -> f32 output [M x 1024]
// Block 256 threads (8 waves), tile 128(M) x 128(N), waves 4x2, 32x64 per wave:
// 8 WMMAs per 12 ds_load_b128 (A-fragments reused across 4 B-tiles).
// Double-buffered LDS, one barrier per K-step; global prefetch 2 steps ahead.
// ---------------------------------------------------------------------------
template <int MODE>
__global__ __launch_bounds__(256)
void gemm_bf16wmma(const float* __restrict__ A,
                   const float* __restrict__ W,
                   const float* __restrict__ bias,
                   __bf16* __restrict__ qws, __bf16* __restrict__ kws,
                   __bf16* __restrict__ vws, float* __restrict__ outf)
{
    __shared__ __attribute__((aligned(32))) __bf16 Asm[2][128][32];
    __shared__ __attribute__((aligned(32))) __bf16 Bsm[2][128][32];

    const int t    = threadIdx.x;
    const int lane = t & 31;
    const int w    = t >> 5;
    const int wm   = w >> 1;     // 0..3  (M)
    const int wn   = w & 1;      // 0..1  (N)
    const int blockN = blockIdx.x * 128;
    const int blockM = blockIdx.y * 128;

    v8f acc[2][4];
#pragma unroll
    for (int i = 0; i < 2; ++i)
#pragma unroll
        for (int j = 0; j < 4; ++j) acc[i][j] = vzero8();

    // staging: 16 contiguous f32 per thread for each of A and B
    const int sr = t >> 1;            // 0..127 (row)
    const int sk = (t & 1) * 16;      // 0,16

    const float* Arow = A + (size_t)(blockM + sr) * 1024 + sk;
    const float* Wrow = W + (size_t)(blockN + sr) * 1024 + sk;

    const int kbase = (lane < 16) ? 0 : 8;   // A-fragment K base
    const int khalf = (lane < 16) ? 0 : 16;  // unused; B-fragments use kbase too

    // ---- prologue: stage K-step 0 into buffer 0 ----
    {
        float4 a0 = *(const float4*)(Arow);
        float4 a1 = *(const float4*)(Arow + 4);
        float4 a2 = *(const float4*)(Arow + 8);
        float4 a3 = *(const float4*)(Arow + 12);
        *(v8bf*)&Asm[0][sr][sk]     = cvt8(a0, a1);
        *(v8bf*)&Asm[0][sr][sk + 8] = cvt8(a2, a3);
        float4 b0 = *(const float4*)(Wrow);
        float4 b1 = *(const float4*)(Wrow + 4);
        float4 b2 = *(const float4*)(Wrow + 8);
        float4 b3 = *(const float4*)(Wrow + 12);
        *(v8bf*)&Bsm[0][sr][sk]     = cvt8(b0, b1);
        *(v8bf*)&Bsm[0][sr][sk + 8] = cvt8(b2, b3);
    }
    __syncthreads();

    for (int step = 0; step < 32; ++step) {
        const int  cur      = step & 1;
        const bool haveNext = (step + 1) < 32;

        // ---- issue next K-step global loads early (hide HBM/L2 latency) ----
        v8bf av0N, av1N, bv0N, bv1N;
        if (haveNext) {
            const int kn = (step + 1) * 32;
            float4 a0 = *(const float4*)(Arow + kn);
            float4 a1 = *(const float4*)(Arow + kn + 4);
            float4 a2 = *(const float4*)(Arow + kn + 8);
            float4 a3 = *(const float4*)(Arow + kn + 12);
            av0N = cvt8(a0, a1);
            av1N = cvt8(a2, a3);
            float4 b0 = *(const float4*)(Wrow + kn);
            float4 b1 = *(const float4*)(Wrow + kn + 4);
            float4 b2 = *(const float4*)(Wrow + kn + 8);
            float4 b3 = *(const float4*)(Wrow + kn + 12);
            bv0N = cvt8(b0, b1);
            bv1N = cvt8(b2, b3);
            if (step + 2 < 32) {                       // gfx1250 global_prefetch_b8
                __builtin_prefetch(Arow + kn + 32, 0, 3);
                __builtin_prefetch(Wrow + kn + 32, 0, 3);
            }
        }

        // ---- fragments + 8 WMMAs from current buffer ----
        BF16x16 afrag[2], bfrag[4];
#pragma unroll
        for (int i = 0; i < 2; ++i) {
            const int m = wm * 32 + i * 16 + (lane & 15);
            afrag[i].h[0] = *(const v8bf*)&Asm[cur][m][kbase];
            afrag[i].h[1] = *(const v8bf*)&Asm[cur][m][kbase + 16];
        }
#pragma unroll
        for (int j = 0; j < 4; ++j) {
            const int n = wn * 64 + j * 16 + (lane & 15);
            bfrag[j].h[0] = *(const v8bf*)&Bsm[cur][n][kbase];
            bfrag[j].h[1] = *(const v8bf*)&Bsm[cur][n][kbase + 16];
        }
#pragma unroll
        for (int i = 0; i < 2; ++i)
#pragma unroll
            for (int j = 0; j < 4; ++j)
                acc[i][j] = __builtin_amdgcn_wmma_f32_16x16x32_bf16(
                    false, afrag[i].v, false, bfrag[j].v,
                    (short)0, acc[i][j], false, false);

        // ---- commit next tile into the other buffer ----
        if (haveNext) {
            *(v8bf*)&Asm[cur ^ 1][sr][sk]     = av0N;
            *(v8bf*)&Asm[cur ^ 1][sr][sk + 8] = av1N;
            *(v8bf*)&Bsm[cur ^ 1][sr][sk]     = bv0N;
            *(v8bf*)&Bsm[cur ^ 1][sr][sk + 8] = bv1N;
        }
        __syncthreads();
    }

    // ---- epilogue ----
#pragma unroll
    for (int i = 0; i < 2; ++i) {
        const int rowBase = blockM + wm * 32 + i * 16 + ((lane < 16) ? 0 : 8);
#pragma unroll
        for (int j = 0; j < 4; ++j) {
            const int colBase = blockN + wn * 64 + j * 16;   // 16-aligned
            const int col = colBase + (lane & 15);
            const float bc = bias[col];
            if (MODE == 0) {
                // 16-wide tile lies inside one head's 64-wide chunk:
                const int sec    = colBase >> 10;            // 0=q 1=k 2=v
                const int dd     = colBase & 1023;
                const int h      = dd >> 6;
                const int hd     = (dd & 63) + (lane & 15);
                __bf16* dst = (sec == 0) ? qws : ((sec == 1) ? kws : vws);
                const float scale = (sec == 0) ? 0.125f : 1.0f;  // 1/sqrt(64)
#pragma unroll
                for (int r = 0; r < 8; ++r) {
                    const int row = rowBase + r;
                    const int bb = row >> 11, ss = row & 2047;
                    const size_t idx =
                        (((size_t)(bb * 16 + h) * 2048) + ss) * 64 + hd;
                    dst[idx] = (__bf16)((acc[i][j][r] + bc) * scale);
                }
            } else {
#pragma unroll
                for (int r = 0; r < 8; ++r) {
                    const int row = rowBase + r;
                    outf[(size_t)row * 1024 + col] = acc[i][j][r] + bc;
                }
            }
        }
    }
}

// ---------------------------------------------------------------------------
// Flash attention: block = 4 waves = 64 query rows of one (b,h).
// KV loop in blocks of 64; K tile staged by the Tensor Data Mover (TDM) when
// available (TENSORcnt + s_wait_tensorcnt), V staged transposed manually.
// Dynamic LDS layout: K[64][64] @0, V[64][80] @8192, P[4][16][64] @18432.
// ---------------------------------------------------------------------------
#define ATT_SMEM_BYTES (8192 + 10240 + 8192)

__global__ __launch_bounds__(128)
void flash_attn(const __bf16* __restrict__ qws, const __bf16* __restrict__ kws,
                const __bf16* __restrict__ vws, float* __restrict__ ctx)
{
    extern __shared__ char smem[];
    __bf16 (*Ksm)[64]     = (__bf16(*)[64])(smem);            // [kv][hd]
    __bf16 (*Vsm)[80]     = (__bf16(*)[80])(smem + 8192);     // [hd][kv] (+pad)
    __bf16 (*Psm)[16][64] = (__bf16(*)[16][64])(smem + 18432);

    const int t    = threadIdx.x;
    const int lane = t & 31;
    const int w    = t >> 5;              // 0..3
    const int bh   = blockIdx.y;          // b*16 + h
    const int sTile = blockIdx.x * 64;
    const size_t headBase = (size_t)bh * 2048 * 64;

    const int kbase   = (lane < 16) ? 0 : 8;   // A-frag
    const int khalf16 = (lane < 16) ? 0 : 16;  // unused; B-frags use kbase too

    // ---- load Q fragments once (Q already scaled by 1/8) ----
    BF16x16 qf[2];
    {
        const int sq = sTile + w * 16 + (lane & 15);
        const __bf16* qrow = qws + headBase + (size_t)sq * 64;
#pragma unroll
        for (int c = 0; c < 2; ++c) {
            qf[c].h[0] = *(const v8bf*)(qrow + c * 32 + kbase);
            qf[c].h[1] = *(const v8bf*)(qrow + c * 32 + kbase + 16);
        }
    }

    v8f o[4];
#pragma unroll
    for (int j = 0; j < 4; ++j) o[j] = vzero8();
    float mrun[8], lrun[8];
#pragma unroll
    for (int r = 0; r < 8; ++r) { mrun[r] = -1e30f; lrun[r] = 0.f; }

    const int stR = t >> 1;         // 0..63 staging kv row
    const int stH = (t & 1) * 32;   // half of hd

    for (int kv0 = 0; kv0 < 2048; kv0 += 64) {
#if USE_TDM
        // ---- K tile via Tensor Data Mover: 64x64 bf16 -> LDS offset 0 ----
        if (w == 0) {
            const unsigned long long gaddr =
                (unsigned long long)(uintptr_t)(kws + headBase + (size_t)kv0 * 64);
            u32x4 g0;
            g0[0] = 1u;                                        // count=1
            g0[1] = 0u;                                        // lds_addr = 0 (Ksm)
            g0[2] = (unsigned)(gaddr & 0xffffffffu);
            g0[3] = (unsigned)((gaddr >> 32) & 0x1ffffffu) | (2u << 30); // type=2
            i32x8 g1;
            g1[0] = (int)(1u << 16);       // data_size = 2 bytes
            g1[1] = (int)(64u << 16);      // tensor_dim0 = 64
            g1[2] = (int)(64u << 16);      // tensor_dim1 = 64
            g1[3] = (int)(64u << 16);      // tile_dim0 = 64
            g1[4] = 64;                    // tile_dim1 = 64
            g1[5] = 64;                    // tensor_dim0_stride = 64
            g1[6] = 0;
            g1[7] = 0;
            i32x4 z4 = {0, 0, 0, 0};
            __builtin_amdgcn_tensor_load_to_lds(g0, g1, z4, z4, 0);
        }
#else
        // ---- K tile staged manually ----
        {
            const __bf16* krow = kws + headBase + (size_t)(kv0 + stR) * 64 + stH;
            v8bf k0 = ((const v8bf*)krow)[0];
            v8bf k1 = ((const v8bf*)krow)[1];
            v8bf k2 = ((const v8bf*)krow)[2];
            v8bf k3 = ((const v8bf*)krow)[3];
            *(v8bf*)&Ksm[stR][stH]      = k0;
            *(v8bf*)&Ksm[stR][stH + 8]  = k1;
            *(v8bf*)&Ksm[stR][stH + 16] = k2;
            *(v8bf*)&Ksm[stR][stH + 24] = k3;
        }
#endif
        // ---- V tile staged transposed ----
        {
            const __bf16* vrow = vws + headBase + (size_t)(kv0 + stR) * 64 + stH;
            v8bf tv0 = ((const v8bf*)vrow)[0];
            v8bf tv1 = ((const v8bf*)vrow)[1];
            v8bf tv2 = ((const v8bf*)vrow)[2];
            v8bf tv3 = ((const v8bf*)vrow)[3];
#pragma unroll
            for (int e = 0; e < 8; ++e) {
                Vsm[stH + e][stR]      = tv0[e];
                Vsm[stH + 8 + e][stR]  = tv1[e];
                Vsm[stH + 16 + e][stR] = tv2[e];
                Vsm[stH + 24 + e][stR] = tv3[e];
            }
        }
        if (kv0 + 64 < 2048) {
            __builtin_prefetch(kws + headBase + (size_t)(kv0 + 64 + stR) * 64 + stH, 0, 3);
            __builtin_prefetch(vws + headBase + (size_t)(kv0 + 64 + stR) * 64 + stH, 0, 3);
        }
#if USE_TDM
        if (w == 0) __builtin_amdgcn_s_wait_tensorcnt(0);
#endif
        __syncthreads();

        // ---- scores: 16x64 per wave, K-dim = HD = 64 (2 WMMA chunks) ----
        v8f sc[4];
#pragma unroll
        for (int j = 0; j < 4; ++j) sc[j] = vzero8();
#pragma unroll
        for (int j = 0; j < 4; ++j) {
            const int kvl = j * 16 + (lane & 15);
#pragma unroll
            for (int c = 0; c < 2; ++c) {
                BF16x16 bk;
                bk.h[0] = *(const v8bf*)&Ksm[kvl][c * 32 + kbase];
                bk.h[1] = *(const v8bf*)&Ksm[kvl][c * 32 + kbase + 16];
                sc[j] = __builtin_amdgcn_wmma_f32_16x16x32_bf16(
                    false, qf[c].v, false, bk.v, (short)0, sc[j], false, false);
            }
        }

        // ---- online softmax (rows striped over 16-lane halves) ----
        const int rowM = (lane < 16) ? 0 : 8;
#pragma unroll
        for (int r = 0; r < 8; ++r) {
            float mx = fmaxf(fmaxf(sc[0][r], sc[1][r]), fmaxf(sc[2][r], sc[3][r]));
#pragma unroll
            for (int msk = 1; msk < 16; msk <<= 1)
                mx = fmaxf(mx, __shfl_xor(mx, msk, 32));
            const float mnew = fmaxf(mrun[r], mx);
            const float corr = exp2f((mrun[r] - mnew) * L2E);
            lrun[r] *= corr;
#pragma unroll
            for (int j = 0; j < 4; ++j) o[j][r] *= corr;
            float sum = 0.f;
#pragma unroll
            for (int j = 0; j < 4; ++j) {
                const float p = exp2f((sc[j][r] - mnew) * L2E);
                sum += p;
                Psm[w][rowM + r][j * 16 + (lane & 15)] = (__bf16)p;
            }
#pragma unroll
            for (int msk = 1; msk < 16; msk <<= 1)
                sum += __shfl_xor(sum, msk, 32);
            lrun[r] += sum;
            mrun[r] = mnew;
        }

        // ---- O += P(16x64) x V(64x64) ----
        BF16x16 pf[2];
        {
            const __bf16* prow = &Psm[w][lane & 15][0];
#pragma unroll
            for (int c = 0; c < 2; ++c) {
                pf[c].h[0] = *(const v8bf*)(prow + c * 32 + kbase);
                pf[c].h[1] = *(const v8bf*)(prow + c * 32 + kbase + 16);
            }
        }
#pragma unroll
        for (int j2 = 0; j2 < 4; ++j2) {
            const int hdl = j2 * 16 + (lane & 15);
#pragma unroll
            for (int c = 0; c < 2; ++c) {
                BF16x16 vf;
                vf.h[0] = *(const v8bf*)&Vsm[hdl][c * 32 + kbase];
                vf.h[1] = *(const v8bf*)&Vsm[hdl][c * 32 + kbase + 16];
                o[j2] = __builtin_amdgcn_wmma_f32_16x16x32_bf16(
                    false, pf[c].v, false, vf.v, (short)0, o[j2], false, false);
            }
        }
        __syncthreads();
    }

    // ---- normalize + write ctx f32 [B,S,D] ----
    const int b = bh >> 4, h = bh & 15;
#pragma unroll
    for (int r = 0; r < 8; ++r) {
        const float inv = 1.0f / lrun[r];
        const int ss = sTile + w * 16 + r + ((lane < 16) ? 0 : 8);
#pragma unroll
        for (int j = 0; j < 4; ++j) {
            const int d = h * 64 + j * 16 + (lane & 15);
            ctx[((size_t)b * 2048 + ss) * 1024 + d] = o[j][r] * inv;
        }
    }
}

// ---------------------------------------------------------------------------
// Launch
// ---------------------------------------------------------------------------
extern "C" void kernel_launch(void* const* d_in, const int* in_sizes, int n_in,
                              void* d_out, int out_size, void* d_ws, size_t ws_size,
                              hipStream_t stream)
{
    const float* x      = (const float*)d_in[0];
    const float* qkv_w  = (const float*)d_in[1];
    const float* qkv_b  = (const float*)d_in[2];
    const float* proj_w = (const float*)d_in[3];
    const float* proj_b = (const float*)d_in[4];
    float* out = (float*)d_out;

    // workspace: Q/K/V bf16 (8 MB each) + ctx f32 (16 MB) = 40 MB
    char* ws = (char*)d_ws;
    __bf16* qws = (__bf16*)(ws);
    __bf16* kws = (__bf16*)(ws + (size_t)(8u << 20));
    __bf16* vws = (__bf16*)(ws + (size_t)(16u << 20));
    float*  ctx = (float*) (ws + (size_t)(24u << 20));

    // 1) fused QKV projection: [4096 x 3072]
    dim3 g1(3072 / 128, 4096 / 128);
    gemm_bf16wmma<0><<<g1, 256, 0, stream>>>(x, qkv_w, qkv_b,
                                             qws, kws, vws, nullptr);

    // 2) flash attention: grid = (S/64, B*H)
    dim3 g2(2048 / 64, 32);
    flash_attn<<<g2, 128, ATT_SMEM_BYTES, stream>>>(qws, kws, vws, ctx);

    // 3) output projection: [4096 x 1024]
    dim3 g3(1024 / 128, 4096 / 128);
    gemm_bf16wmma<1><<<g3, 256, 0, stream>>>(ctx, proj_w, proj_b,
                                             nullptr, nullptr, nullptr, out);
}
